// GClip_2817498546750
// MI455X (gfx1250) — compile-verified, matrix operand order checked by reading, not run on hardware
//
#include <hip/hip_runtime.h>
#include <hip/hip_bf16.h>
#include <math.h>

// ---------------------------------------------------------------------------
// GClip GCN/GAE forward for MI455X (gfx1250, wave32, WMMA).
// Dense GEMMs run through v_wmma_f32_16x16x32_bf16 (fp32 accumulate; inputs
// converted fp32->bf16 while staging tiles into LDS). SpMM is edge-parallel
// atomic scatter-add. Everything else is cheap elementwise work.
// ---------------------------------------------------------------------------

typedef __attribute__((ext_vector_type(16))) __bf16 v16bf;
typedef __attribute__((ext_vector_type(8)))  __bf16 bf16x8;
typedef __attribute__((ext_vector_type(8)))  float  v8f;

__device__ __forceinline__ unsigned short f2bf(float f) {
  unsigned u = __builtin_bit_cast(unsigned, f);
  unsigned r = u + 0x7FFFu + ((u >> 16) & 1u);   // round-to-nearest-even
  return (unsigned short)(r >> 16);
}

// ---------------------------------------------------------------------------
// Tiled GEMM: C[M,N] = A[M,K] (row-major, fp32) x B (fp32)
//   TRANSB=0 : B is [K,N] row-major
//   TRANSB=1 : B is [N,K] row-major (i.e. C = A * B^T; used for h @ h^T)
// EPI: 0 = none, 1 = +bias[N], 2 = sigmoid
// Block tile 64(M) x 256(N), K-step 32. 8 wave32 waves in a 2x4 grid; each
// wave owns a 32x64 tile -> 2x4 WMMA fragments = 8 v_wmma per K-step.
// Interior tiles take a uniform guard-free staging path (float4 loads).
// ---------------------------------------------------------------------------
#define BM  64
#define BN  256
#define BK  32
#define BKP 40   // padded row length (bf16 elements) to dodge bank conflicts

template <int EPI, bool TRANSB>
__global__ __launch_bounds__(256)
void gemm_wmma_k(const float* __restrict__ A, const float* __restrict__ B,
                 const float* __restrict__ bias, float* __restrict__ C,
                 int M, int N, int K) {
  __shared__ __align__(16) unsigned short As[BM][BKP];
  __shared__ __align__(16) unsigned short Bs[BN][BKP];

  const int tid   = threadIdx.x;
  const int mBase = blockIdx.y * BM;
  const int nBase = blockIdx.x * BN;

  const int lane = tid & 31;
  const int wave = tid >> 5;
  const int wm   = wave >> 2;        // 0..1  -> m offset wm*32
  const int wn   = wave & 3;         // 0..3  -> n offset wn*64
  const int l16  = lane & 15;
  const int half = lane >> 4;
  const int k0   = half << 3;        // 0 or 8
  const int k1   = 16 + (half << 3); // 16 or 24

  // uniform fast-path eligibility (alignment of all our shapes is multiple-of-4)
  const bool alignOK = TRANSB ? ((K & 3) == 0) : (((K & 3) == 0) && ((N & 3) == 0));
  const bool fullMN  = (mBase + BM <= M) && (nBase + BN <= N) && alignOK;

  v8f acc[2][4] = {};

  for (int kb = 0; kb < K; kb += BK) {
    const bool fullK = (kb + BK <= K);

    if (fullMN && fullK) {
      // ---------------- fast staging: no per-element guards ----------------
      // A tile: 64x32, 512 float4 loads -> 2 per thread
#pragma unroll
      for (int i = 0; i < 2; ++i) {
        int idx = tid + i * 256;
        int r   = idx >> 3;
        int c4  = (idx & 7) << 2;
        const float* pa = A + (size_t)(mBase + r) * K + kb + c4;
        if (i == 0 && kb + 2 * BK <= K)
          __builtin_prefetch((const void*)(pa + BK), 0, 3);  // next K tile
        const float4 v = *(const float4*)pa;
        As[r][c4 + 0] = f2bf(v.x); As[r][c4 + 1] = f2bf(v.y);
        As[r][c4 + 2] = f2bf(v.z); As[r][c4 + 3] = f2bf(v.w);
      }
      if (TRANSB) {
        // B' = [N,K]; 256x32 tile, 2048 float4 -> 8 per thread
#pragma unroll
        for (int i = 0; i < 8; ++i) {
          int idx = tid + i * 256;
          int r   = idx >> 3;            // 0..255 (n-local)
          int c4  = (idx & 7) << 2;      // k-local
          const float4 v = *(const float4*)(B + (size_t)(nBase + r) * K + kb + c4);
          Bs[r][c4 + 0] = f2bf(v.x); Bs[r][c4 + 1] = f2bf(v.y);
          Bs[r][c4 + 2] = f2bf(v.z); Bs[r][c4 + 3] = f2bf(v.w);
        }
      } else {
        // B = [K,N]; 32x256 tile, transpose into Bs[n][k]
#pragma unroll
        for (int i = 0; i < 8; ++i) {
          int idx = tid + i * 256;
          int r   = idx >> 6;            // 0..31 (k-local)
          int c4  = (idx & 63) << 2;     // n-local
          const float4 v = *(const float4*)(B + (size_t)(kb + r) * N + nBase + c4);
          Bs[c4 + 0][r] = f2bf(v.x); Bs[c4 + 1][r] = f2bf(v.y);
          Bs[c4 + 2][r] = f2bf(v.z); Bs[c4 + 3][r] = f2bf(v.w);
        }
      }
    } else {
      // ---------------- guarded staging (edge tiles only) ----------------
#pragma unroll
      for (int i = 0; i < 2; ++i) {
        int idx = tid + i * 256;
        int r   = idx >> 3;
        int c4  = (idx & 7) << 2;
        int gm  = mBase + r;
#pragma unroll
        for (int j = 0; j < 4; ++j) {
          int gk  = kb + c4 + j;
          float v = (gm < M && gk < K) ? A[(size_t)gm * K + gk] : 0.0f;
          As[r][c4 + j] = f2bf(v);
        }
      }
      if (TRANSB) {
#pragma unroll
        for (int i = 0; i < 8; ++i) {
          int idx = tid + i * 256;
          int r   = idx >> 3;
          int c4  = (idx & 7) << 2;
          int gn  = nBase + r;
#pragma unroll
          for (int j = 0; j < 4; ++j) {
            int gk  = kb + c4 + j;
            float v = (gn < N && gk < K) ? B[(size_t)gn * K + gk] : 0.0f;
            Bs[r][c4 + j] = f2bf(v);
          }
        }
      } else {
#pragma unroll
        for (int i = 0; i < 8; ++i) {
          int idx = tid + i * 256;
          int r   = idx >> 6;
          int c4  = (idx & 63) << 2;
          int gk  = kb + r;
#pragma unroll
          for (int j = 0; j < 4; ++j) {
            int gn  = nBase + c4 + j;
            float v = (gk < K && gn < N) ? B[(size_t)gk * N + gn] : 0.0f;
            Bs[c4 + j][r] = f2bf(v);
          }
        }
      }
    }
    __syncthreads();

    // ---- WMMA compute: 2x4 fragments per wave ----
    v16bf afrag[2], bfrag[4];
#pragma unroll
    for (int sm = 0; sm < 2; ++sm) {
      int m = wm * 32 + sm * 16 + l16;
      bf16x8 lo = *(const bf16x8*)&As[m][k0];
      bf16x8 hi = *(const bf16x8*)&As[m][k1];
#pragma unroll
      for (int j = 0; j < 8; ++j) { afrag[sm][j] = lo[j]; afrag[sm][8 + j] = hi[j]; }
    }
#pragma unroll
    for (int sn = 0; sn < 4; ++sn) {
      int n = wn * 64 + sn * 16 + l16;
      bf16x8 lo = *(const bf16x8*)&Bs[n][k0];
      bf16x8 hi = *(const bf16x8*)&Bs[n][k1];
#pragma unroll
      for (int j = 0; j < 8; ++j) { bfrag[sn][j] = lo[j]; bfrag[sn][8 + j] = hi[j]; }
    }
#pragma unroll
    for (int sm = 0; sm < 2; ++sm)
#pragma unroll
      for (int sn = 0; sn < 4; ++sn)
        acc[sm][sn] = __builtin_amdgcn_wmma_f32_16x16x32_bf16(
            false, afrag[sm], false, bfrag[sn], (short)0, acc[sm][sn],
            false, false);
    __syncthreads();
  }

  // ---- epilogue ----
  if (fullMN) {
#pragma unroll
    for (int sm = 0; sm < 2; ++sm)
#pragma unroll
      for (int sn = 0; sn < 4; ++sn) {
        int col = nBase + wn * 64 + sn * 16 + l16;
#pragma unroll
        for (int r = 0; r < 8; ++r) {
          int row = mBase + wm * 32 + sm * 16 + (half << 3) + r;
          float v = acc[sm][sn][r];
          if (EPI == 1) v += bias[col];
          if (EPI == 2) v = 1.0f / (1.0f + __expf(-v));
          C[(size_t)row * N + col] = v;
        }
      }
  } else {
#pragma unroll
    for (int sm = 0; sm < 2; ++sm)
#pragma unroll
      for (int sn = 0; sn < 4; ++sn) {
        int col = nBase + wn * 64 + sn * 16 + l16;
#pragma unroll
        for (int r = 0; r < 8; ++r) {
          int row = mBase + wm * 32 + sm * 16 + (half << 3) + r;
          if (row < M && col < N) {
            float v = acc[sm][sn][r];
            if (EPI == 1) v += bias[col];
            if (EPI == 2) v = 1.0f / (1.0f + __expf(-v));
            C[(size_t)row * N + col] = v;
          }
        }
      }
  }
}

// ---------------------------------------------------------------------------
// SpMM: out[rows[e], :] += vals[e] * dense[cols[e], :]   (atomic scatter)
// ---------------------------------------------------------------------------
__global__ void spmm_k(const int* __restrict__ rows, const int* __restrict__ cols,
                       const float* __restrict__ vals, const float* __restrict__ dense,
                       float* __restrict__ out, int E, int D) {
  int t = blockIdx.x * blockDim.x + threadIdx.x;
  int perEdge = D >> 2;
  int e = t / perEdge;
  if (e >= E) return;
  int c = (t - e * perEdge) << 2;
  float v = vals[e];
  const float4 x = *(const float4*)(dense + (size_t)cols[e] * D + c);
  float* o = out + (size_t)rows[e] * D + c;
  atomicAdd(o + 0, v * x.x);
  atomicAdd(o + 1, v * x.y);
  atomicAdd(o + 2, v * x.z);
  atomicAdd(o + 3, v * x.w);
}

__global__ void fill0_k(float* p, long long n) {
  long long i = (long long)blockIdx.x * blockDim.x + threadIdx.x;
  long long stride = (long long)gridDim.x * blockDim.x;
  for (; i < n; i += stride) p[i] = 0.0f;
}

__global__ void bias_relu_k(float* p, const float* __restrict__ bias,
                            long long total, int D) {
  long long i = (long long)blockIdx.x * blockDim.x + threadIdx.x;
  long long stride = (long long)gridDim.x * blockDim.x;
  for (; i < total; i += stride) {
    float v = p[i] + bias[i % D];
    p[i] = v > 0.0f ? v : 0.0f;
  }
}

__global__ void concat_k(const float* __restrict__ a, const float* __restrict__ b,
                         float* __restrict__ o, int M, int D) {
  long long total = (long long)M * 2 * D;
  long long i = (long long)blockIdx.x * blockDim.x + threadIdx.x;
  long long stride = (long long)gridDim.x * blockDim.x;
  for (; i < total; i += stride) {
    long long row = i / (2 * D);
    int col = (int)(i - row * 2 * D);
    o[i] = (col < D) ? a[row * D + col] : b[row * D + col - D];
  }
}

// one block (256 threads) per row; D must equal 256
__global__ void rownorm_k(const float* __restrict__ h, float* __restrict__ out, int D) {
  __shared__ float red[256];
  int row = blockIdx.x;
  int tid = threadIdx.x;
  float v = h[(size_t)row * D + tid];
  red[tid] = v * v;
  __syncthreads();
  for (int s = 128; s > 0; s >>= 1) {
    if (tid < s) red[tid] += red[tid + s];
    __syncthreads();
  }
  float inv = rsqrtf(red[0]);
  out[(size_t)row * D + tid] = v * inv;
}

__global__ void logsoftmax_k(const float* __restrict__ z, float* __restrict__ out,
                             int M, int C) {
  int r = blockIdx.x * blockDim.x + threadIdx.x;
  if (r >= M) return;
  const float* zr = z + (size_t)r * C;
  float mx = zr[0];
  for (int c = 1; c < C; ++c) mx = fmaxf(mx, zr[c]);
  float s = 0.0f;
  for (int c = 0; c < C; ++c) s += expf(zr[c] - mx);
  float lse = mx + logf(s);
  for (int c = 0; c < C; ++c) out[(size_t)r * C + c] = zr[c] - lse;
}

__global__ void expscalar_k(const float* __restrict__ in, float* __restrict__ out) {
  if (blockIdx.x == 0 && threadIdx.x == 0) out[0] = expf(in[0]);
}

// ---------------------------------------------------------------------------
// Host-side orchestration
// ---------------------------------------------------------------------------
static inline void launch_gemm(int epi, bool transb, const float* A, const float* B,
                               const float* bias, float* C, int M, int N, int K,
                               hipStream_t s) {
  dim3 g((N + BN - 1) / BN, (M + BM - 1) / BM), b(256);
  if (transb) {
    if (epi == 2) gemm_wmma_k<2, true><<<g, b, 0, s>>>(A, B, bias, C, M, N, K);
    else          gemm_wmma_k<0, true><<<g, b, 0, s>>>(A, B, bias, C, M, N, K);
  } else {
    if (epi == 1)      gemm_wmma_k<1, false><<<g, b, 0, s>>>(A, B, bias, C, M, N, K);
    else if (epi == 2) gemm_wmma_k<2, false><<<g, b, 0, s>>>(A, B, bias, C, M, N, K);
    else               gemm_wmma_k<0, false><<<g, b, 0, s>>>(A, B, bias, C, M, N, K);
  }
}

static inline void zero_buf(float* p, long long n, hipStream_t s) {
  int blocks = (int)((n + 255 * 4) / (256 * 4));
  if (blocks < 1) blocks = 1;
  if (blocks > 4096) blocks = 4096;
  fill0_k<<<blocks, 256, 0, s>>>(p, n);
}

static inline void run_spmm(const int* rows, const int* cols, const float* vals,
                            const float* dense, float* out, int E, int D,
                            hipStream_t s) {
  long long threads = (long long)E * (D >> 2);
  int blocks = (int)((threads + 255) / 256);
  spmm_k<<<blocks, 256, 0, s>>>(rows, cols, vals, dense, out, E, D);
}

static inline void run_bias_relu(float* p, const float* bias, int M, int D,
                                 hipStream_t s) {
  long long total = (long long)M * D;
  int blocks = (int)((total + 255 * 4) / (256 * 4));
  if (blocks > 4096) blocks = 4096;
  bias_relu_k<<<blocks, 256, 0, s>>>(p, bias, total, D);
}

extern "C" void kernel_launch(void* const* d_in, const int* in_sizes, int n_in,
                              void* d_out, int out_size, void* d_ws, size_t ws_size,
                              hipStream_t stream) {
  (void)in_sizes; (void)n_in; (void)out_size; (void)ws_size;
  const int NN = 8000, EE = 128000, NF = 1000, H1 = 512, H2 = 256, NC = 10;

  const float* x      = (const float*)d_in[0];
  const int*   s_rows = (const int*)d_in[1];
  const int*   s_cols = (const int*)d_in[2];
  const float* s_vals = (const float*)d_in[3];
  const int*   f_rows = (const int*)d_in[4];
  const int*   f_cols = (const int*)d_in[5];
  const float* f_vals = (const float*)d_in[6];
  const float* W1 = (const float*)d_in[7];  const float* b1 = (const float*)d_in[8];
  const float* W2 = (const float*)d_in[9];  const float* b2 = (const float*)d_in[10];
  const float* W3 = (const float*)d_in[11]; const float* b3 = (const float*)d_in[12];
  const float* g1W1 = (const float*)d_in[13]; const float* g1b1 = (const float*)d_in[14];
  const float* g1W2 = (const float*)d_in[15]; const float* g1b2 = (const float*)d_in[16];
  const float* g2W1 = (const float*)d_in[17]; const float* g2b1 = (const float*)d_in[18];
  const float* g2W2 = (const float*)d_in[19]; const float* g2b2 = (const float*)d_in[20];
  const float* mw0 = (const float*)d_in[21];
  const float* mw1 = (const float*)d_in[22]; const float* mb1 = (const float*)d_in[23];
  const float* mw2 = (const float*)d_in[24]; const float* mb2 = (const float*)d_in[25];
  const float* logit_scale = (const float*)d_in[26];

  // output layout (flat, return order)
  float* out     = (float*)d_out;
  float* o_logit = out;                    // [8000,10]
  float* o_A1    = out + 80000LL;          // [8000,8000]
  float* o_A2    = out + 64080000LL;       // [8000,8000]
  float* o_emb1  = out + 128080000LL;      // [8000,256]
  float* o_emb2  = out + 130128000LL;      // [8000,256]
  float* o_ls    = out + 132176000LL;      // scalar
  float* o_smu   = out + 132176001LL;      // [8000,256]
  float* o_slv   = out + 134224001LL;
  float* o_fmu   = out + 136272001LL;
  float* o_flv   = out + 138320001LL;

  // workspace (fp32), ~74 MB total
  float* ws  = (float*)d_ws;
  float* XW  = ws;                 // [8000,512]
  float* HS  = ws + 4096000LL;     // [8000,512]
  float* HF  = ws + 8192000LL;     // [8000,512]
  float* TMP = ws + 12288000LL;    // [8000,256]
  float* Hh1 = ws + 14336000LL;    // [8000,256]
  float* Hh2 = ws + 16384000LL;    // [8000,256]
  float* ZC  = HS;                 // reuse after GAE: [8000,512]
  float* Z1  = HF;                 // [8000,512]
  float* Z2  = TMP;                // [8000,256]
  float* Z3  = XW;                 // [8000,10]

  const long long sz512 = (long long)NN * H1;
  const long long sz256 = (long long)NN * H2;

  // ---- shared first layer: XW = x @ W1 ----
  launch_gemm(0, false, x, W1, nullptr, XW, NN, H1, NF, stream);
  zero_buf(HS, sz512, stream);
  run_spmm(s_rows, s_cols, s_vals, XW, HS, EE, H1, stream);
  run_bias_relu(HS, b1, NN, H1, stream);          // shidden1
  zero_buf(HF, sz512, stream);
  run_spmm(f_rows, f_cols, f_vals, XW, HF, EE, H1, stream);
  run_bias_relu(HF, b1, NN, H1, stream);          // fhidden1

  // ---- smu / slogvar / fmu / flogvar (all use s-adj per reference) ----
  launch_gemm(0, false, HS, W2, nullptr, TMP, NN, H2, H1, stream);
  zero_buf(o_smu, sz256, stream);
  run_spmm(s_rows, s_cols, s_vals, TMP, o_smu, EE, H2, stream);
  run_bias_relu(o_smu, b2, NN, H2, stream);

  launch_gemm(0, false, HS, W3, nullptr, TMP, NN, H2, H1, stream);
  zero_buf(o_slv, sz256, stream);
  run_spmm(s_rows, s_cols, s_vals, TMP, o_slv, EE, H2, stream);
  run_bias_relu(o_slv, b3, NN, H2, stream);

  launch_gemm(0, false, HF, W2, nullptr, TMP, NN, H2, H1, stream);
  zero_buf(o_fmu, sz256, stream);
  run_spmm(s_rows, s_cols, s_vals, TMP, o_fmu, EE, H2, stream);
  run_bias_relu(o_fmu, b2, NN, H2, stream);

  launch_gemm(0, false, HF, W3, nullptr, TMP, NN, H2, H1, stream);
  zero_buf(o_flv, sz256, stream);
  run_spmm(s_rows, s_cols, s_vals, TMP, o_flv, EE, H2, stream);
  run_bias_relu(o_flv, b3, NN, H2, stream);

  // ---- GAE1 (s-adj): h1 ----
  launch_gemm(0, false, x, g1W1, nullptr, XW, NN, H1, NF, stream);
  zero_buf(HS, sz512, stream);
  run_spmm(s_rows, s_cols, s_vals, XW, HS, EE, H1, stream);
  run_bias_relu(HS, g1b1, NN, H1, stream);
  launch_gemm(0, false, HS, g1W2, nullptr, TMP, NN, H2, H1, stream);
  zero_buf(Hh1, sz256, stream);
  run_spmm(s_rows, s_cols, s_vals, TMP, Hh1, EE, H2, stream);
  run_bias_relu(Hh1, g1b2, NN, H2, stream);

  // ---- GAE2 (f-adj): h2 ----
  launch_gemm(0, false, x, g2W1, nullptr, XW, NN, H1, NF, stream);
  zero_buf(HF, sz512, stream);
  run_spmm(f_rows, f_cols, f_vals, XW, HF, EE, H1, stream);
  run_bias_relu(HF, g2b1, NN, H1, stream);
  launch_gemm(0, false, HF, g2W2, nullptr, TMP, NN, H2, H1, stream);
  zero_buf(Hh2, sz256, stream);
  run_spmm(f_rows, f_cols, f_vals, TMP, Hh2, EE, H2, stream);
  run_bias_relu(Hh2, g2b2, NN, H2, stream);

  // ---- A_pred = sigmoid(h h^T) ----
  launch_gemm(2, true, Hh1, Hh1, nullptr, o_A1, NN, NN, H2, stream);
  launch_gemm(2, true, Hh2, Hh2, nullptr, o_A2, NN, NN, H2, stream);

  // ---- row-normalized embeddings ----
  rownorm_k<<<NN, 256, 0, stream>>>(Hh1, o_emb1, H2);
  rownorm_k<<<NN, 256, 0, stream>>>(Hh2, o_emb2, H2);

  // ---- MLP head ----
  {
    int blocks = (int)(((long long)NN * H1 + 255 * 4) / (256 * 4));
    if (blocks > 4096) blocks = 4096;
    concat_k<<<blocks, 256, 0, stream>>>(Hh1, Hh2, ZC, NN, H2);
  }
  launch_gemm(0, false, ZC, mw0, nullptr, Z1, NN, H1, H1, stream);   // no bias
  launch_gemm(1, false, Z1, mw1, mb1, Z2, NN, H2, H1, stream);
  launch_gemm(1, false, Z2, mw2, mb2, Z3, NN, NC, H2, stream);
  logsoftmax_k<<<(NN + 255) / 256, 256, 0, stream>>>(Z3, o_logit, NN, NC);

  // ---- scalar ----
  expscalar_k<<<1, 32, 0, stream>>>(logit_scale, o_ls);
}
